// Attentions_9156870275154
// MI455X (gfx1250) — compile-verified, matrix operand order checked by reading, not run on hardware
//
#include <hip/hip_runtime.h>
#include <hip/hip_bf16.h>
#include <math.h>

typedef _Float16 half_t;
typedef __attribute__((ext_vector_type(16))) _Float16 v16h;
typedef __attribute__((ext_vector_type(8)))  _Float16 v8h;
typedef __attribute__((ext_vector_type(8)))  float    v8f;
typedef __attribute__((ext_vector_type(4)))  int      v4i_t;
typedef __attribute__((address_space(1))) v4i_t* g4i_p;   // global v4i*
typedef __attribute__((address_space(3))) v4i_t* l4i_p;   // LDS v4i*

#define WMMA_F16(a, b, c) \
  __builtin_amdgcn_wmma_f32_16x16x32_f16(false, (a), false, (b), (short)0, (c), false, false)

#if defined(__HIP_DEVICE_COMPILE__)
#if __has_builtin(__builtin_amdgcn_global_load_async_to_lds_b128) && \
    __has_builtin(__builtin_amdgcn_s_wait_asynccnt)
#define USE_ASYNC_LDS 1
#endif
#endif

// ---------------------------------------------------------------------------
// Weight transpose + f32->f16:  W[K][N] -> Wt[N][K]
// ---------------------------------------------------------------------------
__global__ __launch_bounds__(256) void k_transpose_w(const float* __restrict__ W,
                                                     half_t* __restrict__ Wt,
                                                     int K, int N) {
  int i = blockIdx.x * 256 + threadIdx.x;
  int total = K * N;
  if (i >= total) return;
  int k = i % K;
  int n = i / K;
  Wt[i] = (half_t)W[(size_t)k * N + n];
}

// ---------------------------------------------------------------------------
// f32 -> f16 convert with zero padding beyond nvalid
// ---------------------------------------------------------------------------
__global__ __launch_bounds__(256) void k_conv16(const float* __restrict__ src,
                                                half_t* __restrict__ dst,
                                                int nvalid, int ntotal) {
  int i = blockIdx.x * 256 + threadIdx.x;
  if (i >= ntotal) return;
  dst[i] = (i < nvalid) ? (half_t)src[i] : (half_t)0.0f;
}

// ---------------------------------------------------------------------------
// GroupNorm: stats per (batch, group) over 32*32*20 = 20480 elements
// ---------------------------------------------------------------------------
__global__ __launch_bounds__(256) void k_gn_stats(const float* __restrict__ x,
                                                  float* __restrict__ stats) {
  int bg = blockIdx.x;                 // 0..255
  int b = bg >> 5, g = bg & 31;
  __shared__ float s1[256], s2[256];
  int tid = threadIdx.x;
  float sum = 0.f, sq = 0.f;
  for (int i = tid; i < 20480; i += 256) {
    int p = i / 20, j = i % 20;
    float v = x[((size_t)b * 1024 + p) * 640 + g * 20 + j];
    sum += v; sq += v * v;
  }
  s1[tid] = sum; s2[tid] = sq;
  __syncthreads();
  for (int off = 128; off > 0; off >>= 1) {
    if (tid < off) { s1[tid] += s1[tid + off]; s2[tid] += s2[tid + off]; }
    __syncthreads();
  }
  if (tid == 0) {
    float mean = s1[0] / 20480.f;
    float var  = s2[0] / 20480.f - mean * mean;
    stats[bg * 2 + 0] = mean;
    stats[bg * 2 + 1] = rsqrtf(var + 1e-5f);
  }
}

__global__ __launch_bounds__(256) void k_gn_apply(const float* __restrict__ x,
                                                  const float* __restrict__ stats,
                                                  const float* __restrict__ gamma,
                                                  const float* __restrict__ beta,
                                                  half_t* __restrict__ y) {
  int i = blockIdx.x * 256 + threadIdx.x;
  if (i >= 8192 * 640) return;
  int m = i / 640, c = i % 640;
  int b = m >> 10, g = c / 20;
  float mean = stats[(b * 32 + g) * 2 + 0];
  float rstd = stats[(b * 32 + g) * 2 + 1];
  y[i] = (half_t)((x[i] - mean) * rstd * gamma[c] + beta[c]);
}

// ---------------------------------------------------------------------------
// LayerNorm over last dim 640, one row per block, f32 in -> f16 out
// ---------------------------------------------------------------------------
__global__ __launch_bounds__(256) void k_layernorm16(const float* __restrict__ x,
                                                     const float* __restrict__ g,
                                                     const float* __restrict__ be,
                                                     half_t* __restrict__ y) {
  int row = blockIdx.x;
  int tid = threadIdx.x;
  __shared__ float s1[256], s2[256];
  __shared__ float sMean, sRstd;
  const float* xr = x + (size_t)row * 640;
  float sum = 0.f, sq = 0.f;
  for (int c = tid; c < 640; c += 256) { float v = xr[c]; sum += v; sq += v * v; }
  s1[tid] = sum; s2[tid] = sq;
  __syncthreads();
  for (int off = 128; off > 0; off >>= 1) {
    if (tid < off) { s1[tid] += s1[tid + off]; s2[tid] += s2[tid + off]; }
    __syncthreads();
  }
  if (tid == 0) {
    float mean = s1[0] / 640.f;
    float var  = s2[0] / 640.f - mean * mean;
    sMean = mean; sRstd = rsqrtf(var + 1e-5f);
  }
  __syncthreads();
  float mean = sMean, rstd = sRstd;
  half_t* yr = y + (size_t)row * 640;
  for (int c = tid; c < 640; c += 256)
    yr[c] = (half_t)((xr[c] - mean) * rstd * g[c] + be[c]);
}

// ---------------------------------------------------------------------------
// WMMA GEMM with shared double-buffered LDS B-tile + async global->LDS.
// Block = 8 waves computes 128(M) x 64(N); wave w -> m-tile blockIdx.y*8+w.
// B tile (64 cols x 64 k, 8KB) staged once per block per k-chunk -> 8x reuse.
// out[M][N] = A[M][K](f16,lda) @ Wt[N][K](f16) + bias + resid (resid may = out).
// N multiple of 64, K multiple of 64.
// ---------------------------------------------------------------------------
__global__ __launch_bounds__(256) void k_gemm_wmma(const half_t* __restrict__ A, int lda,
                                                   const half_t* __restrict__ Wt,
                                                   const float* bias,
                                                   const float* resid,
                                                   float* out,
                                                   int M, int N, int K) {
  __shared__ __align__(64) half_t Bs[2][64 * 64];   // [buf][col*64 + k], 16 KB
  int tid = threadIdx.x;
  int wave = tid >> 5;
  int lane = tid & 31;
  int mTiles = (M + 15) >> 4;
  int nc = blockIdx.x;
  int mt = blockIdx.y * 8 + wave;
  if (mt >= mTiles) mt = mTiles - 1;      // keep wave alive for barriers; stores guarded
  int n0 = nc * 64;
  int ncol = lane & 15;
  int kb = (lane >> 4) * 16;
  int arow = mt * 16 + (lane & 15);
  if (arow >= M) arow = M - 1;
  const half_t* aG = A + (size_t)arow * lda + kb;

  // cooperative B staging: thread -> (col = tid>>2, koff = (tid&3)*16)
  int colL = tid >> 2;
  int koff = (tid & 3) << 4;
  const half_t* bG = Wt + (size_t)(n0 + colL) * K + koff;
  half_t* bS = &Bs[0][0] + colL * 64 + koff;

  auto stage = [&](int buf, int kc) {
    const half_t* src = bG + kc;
    half_t* dst = bS + buf * (64 * 64);
#ifdef USE_ASYNC_LDS
    __builtin_amdgcn_global_load_async_to_lds_b128(
        (g4i_p)(void*)src, (l4i_p)(void*)dst, 0, 0);
    __builtin_amdgcn_global_load_async_to_lds_b128(
        (g4i_p)(void*)(src + 8), (l4i_p)(void*)(dst + 8), 0, 0);
#else
    v16h tmp = *(const v16h*)src;
    *(v16h*)dst = tmp;
#endif
  };

  v8f acc[4] = {};
  int nIter = K >> 6;
  stage(0, 0);
  for (int i = 0; i < nIter; ++i) {
    int kc = i << 6;
    bool hasNext = (i + 1) < nIter;
    if (hasNext) stage((i + 1) & 1, kc + 64);
#ifdef USE_ASYNC_LDS
    if (hasNext) __builtin_amdgcn_s_wait_asynccnt(2);  // oldest stage complete
    else         __builtin_amdgcn_s_wait_asynccnt(0);
#endif
    __syncthreads();                                    // buf[i&1] visible to all waves
    const half_t* bbase = &Bs[i & 1][0];
    v16h a0 = *(const v16h*)(aG + kc);
    v16h a1 = *(const v16h*)(aG + kc + 32);
    v16h b0[4], b1[4];
#pragma unroll
    for (int s = 0; s < 4; ++s)
      b0[s] = *(const v16h*)(bbase + (s * 16 + ncol) * 64 + kb);
#pragma unroll
    for (int s = 0; s < 4; ++s)
      b1[s] = *(const v16h*)(bbase + (s * 16 + ncol) * 64 + 32 + kb);
#pragma unroll
    for (int s = 0; s < 4; ++s)
      acc[s] = WMMA_F16(a0, b0[s], acc[s]);
#pragma unroll
    for (int s = 0; s < 4; ++s)
      acc[s] = WMMA_F16(a1, b1[s], acc[s]);
    __syncthreads();                                    // reads done before re-stage
  }

  int mBase = mt * 16 + (lane >> 4) * 8;
#pragma unroll
  for (int s = 0; s < 4; ++s) {
    int col = n0 + s * 16 + ncol;
    float bb = bias ? bias[col] : 0.0f;
#pragma unroll
    for (int r = 0; r < 8; ++r) {
      int m = mBase + r;
      if (m < M) {
        size_t idx = (size_t)m * N + col;
        float v = acc[s][r] + bb;
        if (resid) v += resid[idx];
        out[idx] = v;
      }
    }
  }
}

// ---------------------------------------------------------------------------
// GEGLU fused GEMM with async-LDS double-buffered B (both halves).
// Block = 8 waves computes 128(M) x 64(N of 2560); stages xh cols (n) and
// gate cols (n+2560) B-tiles: 2 x 8KB per 64-k chunk, double buffered (32KB).
// p = A[8192][640] @ ff1 + b; G = xh * gelu(gate), f16 [8192][2560].
// ---------------------------------------------------------------------------
__global__ __launch_bounds__(256) void k_gemm_geglu(const half_t* __restrict__ A,
                                                    const half_t* __restrict__ Wt,  // [5120][640]
                                                    const float* __restrict__ bias, // [5120]
                                                    half_t* __restrict__ G) {
  __shared__ __align__(64) half_t Bs[2][2][64 * 64];  // [buf][half][col*64+k], 32 KB
  const int K = 640;
  int tid = threadIdx.x;
  int wave = tid >> 5;
  int lane = tid & 31;
  int nc = blockIdx.x;                   // 0..39
  int mt = blockIdx.y * 8 + wave;        // 0..511
  int n0 = nc * 64;
  int ncol = lane & 15;
  int kb = (lane >> 4) * 16;
  const half_t* aG = A + (size_t)(mt * 16 + (lane & 15)) * K + kb;

  int colL = tid >> 2;
  int koff = (tid & 3) << 4;
  const half_t* bG1 = Wt + (size_t)(n0 + colL) * K + koff;          // xh cols
  const half_t* bG2 = Wt + (size_t)(n0 + colL + 2560) * K + koff;   // gate cols
  half_t* bS = &Bs[0][0][0] + colL * 64 + koff;

  auto stage = [&](int buf, int kc) {
    const half_t* s1 = bG1 + kc;
    const half_t* s2 = bG2 + kc;
    half_t* d1 = bS + buf * (2 * 64 * 64);
    half_t* d2 = d1 + 64 * 64;
#ifdef USE_ASYNC_LDS
    __builtin_amdgcn_global_load_async_to_lds_b128(
        (g4i_p)(void*)s1, (l4i_p)(void*)d1, 0, 0);
    __builtin_amdgcn_global_load_async_to_lds_b128(
        (g4i_p)(void*)(s1 + 8), (l4i_p)(void*)(d1 + 8), 0, 0);
    __builtin_amdgcn_global_load_async_to_lds_b128(
        (g4i_p)(void*)s2, (l4i_p)(void*)d2, 0, 0);
    __builtin_amdgcn_global_load_async_to_lds_b128(
        (g4i_p)(void*)(s2 + 8), (l4i_p)(void*)(d2 + 8), 0, 0);
#else
    v16h t1 = *(const v16h*)s1;
    *(v16h*)d1 = t1;
    v16h t2 = *(const v16h*)s2;
    *(v16h*)d2 = t2;
#endif
  };

  v8f acc1[4] = {};
  v8f acc2[4] = {};
  const int nIter = K >> 6;              // 10
  stage(0, 0);
  for (int i = 0; i < nIter; ++i) {
    int kc = i << 6;
    bool hasNext = (i + 1) < nIter;
    if (hasNext) stage((i + 1) & 1, kc + 64);
#ifdef USE_ASYNC_LDS
    if (hasNext) __builtin_amdgcn_s_wait_asynccnt(4);
    else         __builtin_amdgcn_s_wait_asynccnt(0);
#endif
    __syncthreads();
    const half_t* h1 = &Bs[i & 1][0][0];
    const half_t* h2 = &Bs[i & 1][1][0];
    v16h a0 = *(const v16h*)(aG + kc);
    v16h a1 = *(const v16h*)(aG + kc + 32);
#pragma unroll
    for (int s = 0; s < 4; ++s) {
      int bo = (s * 16 + ncol) * 64 + kb;
      v16h p0 = *(const v16h*)(h1 + bo);
      v16h q0 = *(const v16h*)(h2 + bo);
      acc1[s] = WMMA_F16(a0, p0, acc1[s]);
      acc2[s] = WMMA_F16(a0, q0, acc2[s]);
    }
#pragma unroll
    for (int s = 0; s < 4; ++s) {
      int bo = (s * 16 + ncol) * 64 + 32 + kb;
      v16h p1 = *(const v16h*)(h1 + bo);
      v16h q1 = *(const v16h*)(h2 + bo);
      acc1[s] = WMMA_F16(a1, p1, acc1[s]);
      acc2[s] = WMMA_F16(a1, q1, acc2[s]);
    }
    __syncthreads();
  }

  int mBase = mt * 16 + (lane >> 4) * 8;
#pragma unroll
  for (int s = 0; s < 4; ++s) {
    int col = n0 + s * 16 + ncol;
    float b1 = bias[col];
    float b2 = bias[col + 2560];
#pragma unroll
    for (int r = 0; r < 8; ++r) {
      int m = mBase + r;
      float xh = acc1[s][r] + b1;
      float gt = acc2[s][r] + b2;
      float th = tanhf(0.7978845608f * gt * (1.0f + 0.044715f * gt * gt));
      float g  = xh * 0.5f * gt * (1.0f + th);
      G[(size_t)m * 2560 + col] = (half_t)g;
    }
  }
}

// ---------------------------------------------------------------------------
// Pack helpers for attention operands (head dim 80 padded to 96)
// ---------------------------------------------------------------------------
__global__ __launch_bounds__(256) void k_pack_qk(const float* __restrict__ src,
                                                 half_t* __restrict__ dst,
                                                 int T, int Tpad) {
  int i = blockIdx.x * 256 + threadIdx.x;
  int total = 64 * Tpad * 96;
  if (i >= total) return;
  int d = i % 96;
  int j = i / 96;
  int t = j % Tpad;
  int bh = j / Tpad;
  int b = bh >> 3, h = bh & 7;
  float v = 0.f;
  if (d < 80 && t < T) v = src[((size_t)b * T + t) * 640 + h * 80 + d];
  dst[i] = (half_t)v;
}

__global__ __launch_bounds__(256) void k_pack_vt(const float* __restrict__ src,
                                                 half_t* __restrict__ dst,
                                                 int T, int Tpad) {
  int i = blockIdx.x * 256 + threadIdx.x;
  int total = 64 * 80 * Tpad;
  if (i >= total) return;
  int t = i % Tpad;
  int j = i / Tpad;
  int d = j % 80;
  int bh = j / 80;
  int b = bh >> 3, h = bh & 7;
  float v = (t < T) ? src[((size_t)b * T + t) * 640 + h * 80 + d] : 0.f;
  dst[i] = (half_t)v;
}

// ---------------------------------------------------------------------------
// Attention core: per (bh, 16-q-row block).  8 waves / 256 threads.
//   Phase 0: stage Q tile (16x96, 3KB) into LDS (shared by all 8 waves)
//   Phase 1: S = scale * Q @ K^T (WMMA), f16 scores -> LDS
//   Phase 2: row softmax in LDS (16 threads per row)
//   Phase 3: O = P @ V^T (WMMA, A-frags from LDS, Vt gives contiguous B-frags)
// ---------------------------------------------------------------------------
__global__ __launch_bounds__(256) void k_attn_wmma(const half_t* __restrict__ Q,
                                                   const half_t* __restrict__ Kt,
                                                   const half_t* __restrict__ Vt,
                                                   half_t* __restrict__ O,
                                                   int Tpad, int Tvalid, float scale) {
  __shared__ __align__(64) half_t P[16][1024];
  __shared__ __align__(64) half_t Qs[16 * 96];
  __shared__ float red[16][16];
  __shared__ float rowMax[16], rowSum[16];

  int qb = blockIdx.x;           // 0..63
  int bh = blockIdx.y;           // 0..63
  int b = bh >> 3, head = bh & 7;
  const half_t* q  = Q  + (size_t)bh * 1024 * 96 + (size_t)qb * 16 * 96;
  const half_t* kk = Kt + (size_t)bh * Tpad * 96;
  const half_t* vt = Vt + (size_t)bh * 80 * Tpad;

  int tid = threadIdx.x;
  int wave = tid >> 5;
  int lane = tid & 31;
  int ncol = lane & 15;
  int kb = (lane >> 4) * 16;

  // ---- Phase 0: stage Q tile into LDS (1536 halves, contiguous) ----
  if (tid < 192) {
    v8h t = *(const v8h*)(q + tid * 8);
    *(v8h*)(&Qs[0] + tid * 8) = t;
  }
  __syncthreads();

  // ---- Phase 1: score tiles ----
  int nT = Tpad >> 4;
  const half_t* aQ = &Qs[0] + (lane & 15) * 96 + kb;
  for (int tt = wave; tt < nT; tt += 8) {
    const half_t* bPtr = kk + (size_t)(tt * 16 + ncol) * 96 + kb;
    v8f acc = {};
#pragma unroll
    for (int d0 = 0; d0 < 96; d0 += 32) {
      v16h a = *(const v16h*)(aQ + d0);
      v16h bf = *(const v16h*)(bPtr + d0);
      acc = WMMA_F16(a, bf, acc);
    }
    int col = tt * 16 + ncol;
    int mb = (lane >> 4) * 8;
    bool valid = col < Tvalid;
#pragma unroll
    for (int r = 0; r < 8; ++r)
      P[mb + r][col] = valid ? (half_t)(acc[r] * scale) : (half_t)(-60000.0f);
  }
  __syncthreads();

  // ---- Phase 2: softmax, 16 threads per row ----
  {
    int r = tid >> 4;
    int sub = tid & 15;
    float mx = -1e30f;
    for (int c = sub; c < Tpad; c += 16) mx = fmaxf(mx, (float)P[r][c]);
    red[r][sub] = mx;
    __syncthreads();
    if (sub == 0) {
      float m2 = red[r][0];
      for (int i = 1; i < 16; ++i) m2 = fmaxf(m2, red[r][i]);
      rowMax[r] = m2;
    }
    __syncthreads();
    float rm = rowMax[r];
    float sm = 0.f;
    for (int c = sub; c < Tpad; c += 16) sm += __expf((float)P[r][c] - rm);
    red[r][sub] = sm;
    __syncthreads();
    if (sub == 0) {
      float s2 = 0.f;
      for (int i = 0; i < 16; ++i) s2 += red[r][i];
      rowSum[r] = s2;
    }
    __syncthreads();
    float inv = 1.0f / rowSum[r];
    for (int c = sub; c < Tpad; c += 16)
      P[r][c] = (half_t)(__expf((float)P[r][c] - rm) * inv);
  }
  __syncthreads();

  // ---- Phase 3: O = P @ V^T ----
  for (int nt = wave; nt < 5; nt += 8) {
    const half_t* bBase = vt + (size_t)(nt * 16 + ncol) * Tpad + kb;
    int arow = lane & 15;
    v8f acc = {};
    for (int k0 = 0; k0 < Tpad; k0 += 32) {
      v16h a = *(const v16h*)(&P[arow][k0 + kb]);
      v16h bf = *(const v16h*)(bBase + k0);
      acc = WMMA_F16(a, bf, acc);
    }
    int mb = (lane >> 4) * 8;
#pragma unroll
    for (int r = 0; r < 8; ++r) {
      int srow = qb * 16 + mb + r;
      O[((size_t)b * 1024 + srow) * 640 + head * 80 + nt * 16 + ncol] = (half_t)acc[r];
    }
  }
}

// ---------------------------------------------------------------------------
// Host orchestration
// ---------------------------------------------------------------------------
extern "C" void kernel_launch(void* const* d_in, const int* in_sizes, int n_in,
                              void* d_out, int out_size, void* d_ws, size_t ws_size,
                              hipStream_t stream) {
  const float* x          = (const float*)d_in[0];
  const float* ctx        = (const float*)d_in[1];
  const float* gn_gamma   = (const float*)d_in[2];
  const float* gn_beta    = (const float*)d_in[3];
  const float* proj_in_w  = (const float*)d_in[4];
  const float* proj_in_b  = (const float*)d_in[5];
  const float* ln1_g      = (const float*)d_in[6];
  const float* ln1_b      = (const float*)d_in[7];
  const float* a1_q       = (const float*)d_in[8];
  const float* a1_k       = (const float*)d_in[9];
  const float* a1_v       = (const float*)d_in[10];
  const float* a1_o       = (const float*)d_in[11];
  const float* a1_ob      = (const float*)d_in[12];
  const float* ln2_g      = (const float*)d_in[13];
  const float* ln2_b      = (const float*)d_in[14];
  const float* a2_q       = (const float*)d_in[15];
  const float* a2_k       = (const float*)d_in[16];
  const float* a2_v       = (const float*)d_in[17];
  const float* a2_o       = (const float*)d_in[18];
  const float* a2_ob      = (const float*)d_in[19];
  const float* ln3_g      = (const float*)d_in[20];
  const float* ln3_b      = (const float*)d_in[21];
  const float* ff1_w      = (const float*)d_in[22];
  const float* ff1_b      = (const float*)d_in[23];
  const float* ff2_w      = (const float*)d_in[24];
  const float* ff2_b      = (const float*)d_in[25];
  const float* proj_out_w = (const float*)d_in[26];
  const float* proj_out_b = (const float*)d_in[27];
  float* out = (float*)d_out;

  char* base = (char*)d_ws;
  size_t off = 0;
  auto alloc = [&](size_t bytes) -> char* {
    char* p = base + off;
    off += (bytes + 255) & ~(size_t)255;
    return p;
  };
  half_t* wt_pi  = (half_t*)alloc((size_t)640 * 640 * 2);
  half_t* wt_a1q = (half_t*)alloc((size_t)640 * 640 * 2);
  half_t* wt_a1k = (half_t*)alloc((size_t)640 * 640 * 2);
  half_t* wt_a1v = (half_t*)alloc((size_t)640 * 640 * 2);
  half_t* wt_a1o = (half_t*)alloc((size_t)640 * 640 * 2);
  half_t* wt_a2q = (half_t*)alloc((size_t)640 * 640 * 2);
  half_t* wt_a2k = (half_t*)alloc((size_t)640 * 768 * 2);
  half_t* wt_a2v = (half_t*)alloc((size_t)640 * 768 * 2);
  half_t* wt_a2o = (half_t*)alloc((size_t)640 * 640 * 2);
  half_t* wt_ff1 = (half_t*)alloc((size_t)5120 * 640 * 2);
  half_t* wt_ff2 = (half_t*)alloc((size_t)640 * 2560 * 2);
  half_t* wt_po  = (half_t*)alloc((size_t)640 * 640 * 2);
  float*  stats  = (float*)alloc((size_t)256 * 2 * 4);
  float*  resid  = (float*)alloc((size_t)8192 * 640 * 4);
  float*  tmp32  = (float*)alloc((size_t)8192 * 640 * 4);
  half_t* ln16   = (half_t*)alloc((size_t)8192 * 640 * 2);
  half_t* h_a    = (half_t*)alloc((size_t)8192 * 640 * 2);
  half_t* ctx16  = (half_t*)alloc((size_t)624 * 768 * 2);
  half_t* q16    = (half_t*)alloc((size_t)64 * 1024 * 96 * 2);
  half_t* k16    = (half_t*)alloc((size_t)64 * 1024 * 96 * 2);
  half_t* vt16   = (half_t*)alloc((size_t)64 * 80 * 1024 * 2);
  half_t* g16    = (half_t*)alloc((size_t)8192 * 2560 * 2);
  (void)ws_size; (void)in_sizes; (void)n_in; (void)out_size;

  auto xw = [&](const float* W, half_t* Wt, int K, int N) {
    int total = K * N;
    k_transpose_w<<<(total + 255) / 256, 256, 0, stream>>>(W, Wt, K, N);
  };
  auto gemm = [&](const half_t* A, int lda, const half_t* Wt, const float* bias,
                  const float* rsd, float* o, int M, int N, int K) {
    int mTiles = (M + 15) / 16;
    dim3 grid(N / 64, (mTiles + 7) / 8);
    k_gemm_wmma<<<grid, 256, 0, stream>>>(A, lda, Wt, bias, rsd, o, M, N, K);
  };

  // 0) weights -> f16 transposed
  xw(proj_in_w, wt_pi, 640, 640);
  xw(a1_q, wt_a1q, 640, 640);  xw(a1_k, wt_a1k, 640, 640);
  xw(a1_v, wt_a1v, 640, 640);  xw(a1_o, wt_a1o, 640, 640);
  xw(a2_q, wt_a2q, 640, 640);  xw(a2_k, wt_a2k, 768, 640);
  xw(a2_v, wt_a2v, 768, 640);  xw(a2_o, wt_a2o, 640, 640);
  xw(ff1_w, wt_ff1, 640, 5120);
  xw(ff2_w, wt_ff2, 2560, 640);
  xw(proj_out_w, wt_po, 640, 640);

  const float scale = 0.11180339887498949f;  // 80^-0.5
  const int EL = 8192 * 640;

  // 1) GroupNorm -> f16
  k_gn_stats<<<256, 256, 0, stream>>>(x, stats);
  k_gn_apply<<<(EL + 255) / 256, 256, 0, stream>>>(x, stats, gn_gamma, gn_beta, h_a);

  // 2) proj_in
  gemm(h_a, 640, wt_pi, proj_in_b, nullptr, resid, 8192, 640, 640);

  // 3) self-attention
  k_layernorm16<<<8192, 256, 0, stream>>>(resid, ln1_g, ln1_b, ln16);
  gemm(ln16, 640, wt_a1q, nullptr, nullptr, tmp32, 8192, 640, 640);
  k_pack_qk<<<(64 * 1024 * 96 + 255) / 256, 256, 0, stream>>>(tmp32, q16, 1024, 1024);
  gemm(ln16, 640, wt_a1k, nullptr, nullptr, tmp32, 8192, 640, 640);
  k_pack_qk<<<(64 * 1024 * 96 + 255) / 256, 256, 0, stream>>>(tmp32, k16, 1024, 1024);
  gemm(ln16, 640, wt_a1v, nullptr, nullptr, tmp32, 8192, 640, 640);
  k_pack_vt<<<(64 * 80 * 1024 + 255) / 256, 256, 0, stream>>>(tmp32, vt16, 1024, 1024);
  k_attn_wmma<<<dim3(64, 64), 256, 0, stream>>>(q16, k16, vt16, h_a, 1024, 1024, scale);
  gemm(h_a, 640, wt_a1o, a1_ob, resid, resid, 8192, 640, 640);

  // 4) cross-attention
  k_layernorm16<<<8192, 256, 0, stream>>>(resid, ln2_g, ln2_b, ln16);
  gemm(ln16, 640, wt_a2q, nullptr, nullptr, tmp32, 8192, 640, 640);
  k_pack_qk<<<(64 * 1024 * 96 + 255) / 256, 256, 0, stream>>>(tmp32, q16, 1024, 1024);
  k_conv16<<<(624 * 768 + 255) / 256, 256, 0, stream>>>(ctx, ctx16, 616 * 768, 624 * 768);
  gemm(ctx16, 768, wt_a2k, nullptr, nullptr, tmp32, 624, 640, 768);
  k_pack_qk<<<(64 * 96 * 96 + 255) / 256, 256, 0, stream>>>(tmp32, k16, 77, 96);
  gemm(ctx16, 768, wt_a2v, nullptr, nullptr, tmp32, 624, 640, 768);
  k_pack_vt<<<(64 * 80 * 96 + 255) / 256, 256, 0, stream>>>(tmp32, vt16, 77, 96);
  k_attn_wmma<<<dim3(64, 64), 256, 0, stream>>>(q16, k16, vt16, h_a, 96, 77, scale);
  gemm(h_a, 640, wt_a2o, a2_ob, resid, resid, 8192, 640, 640);

  // 5) GEGLU FF
  k_layernorm16<<<8192, 256, 0, stream>>>(resid, ln3_g, ln3_b, ln16);
  k_gemm_geglu<<<dim3(40, 64), 256, 0, stream>>>(ln16, wt_ff1, ff1_b, g16);
  gemm(g16, 2560, wt_ff2, ff2_b, resid, resid, 8192, 640, 2560);

  // 6) proj_out + input residual -> d_out
  k_conv16<<<(EL + 255) / 256, 256, 0, stream>>>(resid, ln16, EL, EL);
  gemm(ln16, 640, wt_po, proj_out_b, x, out, 8192, 640, 640);
}
